// GatedGCN_layer_11905649344612
// MI455X (gfx1250) — matrix operand order repaired and checked by reading, not compiled, not run on hardware
//
#include <hip/hip_runtime.h>
#include <hip/hip_bf16.h>

typedef __attribute__((ext_vector_type(16))) __bf16        v16bf;
typedef __attribute__((ext_vector_type(8)))  float         v8f;
typedef __attribute__((ext_vector_type(8)))  unsigned int  v8u;

#define Dk     128
#define NN     50000
#define EE     800000
#define BN_EPS 1e-5f

// ---------------------------------------------------------------- helpers

__device__ __forceinline__ unsigned int f32_to_bf16(float f) {
  unsigned int u = __builtin_bit_cast(unsigned int, f);
  return ((u + 0x7FFFu + ((u >> 16) & 1u)) >> 16) & 0xFFFFu;  // RNE
}

// A-matrix fragment (16x32 bf16) per ISA layout:
// lanes 0-15: m=lane, v0-3 -> K=2v+h, v4-7 -> K=16+2(v-4)+h
// lanes 16-31: same m, K+8 / K+24 halves.
__device__ __forceinline__ v16bf load_afrag(const float* __restrict__ X,
                                            int m, int kt, int hi) {
  v8u au;
#pragma unroll
  for (int v = 0; v < 8; ++v) {
    int k0 = kt * 32 + ((v < 4) ? 0 : 16) + hi * 8 + 2 * (v & 3);
    const float* p = X + (size_t)m * Dk + k0;
    au[v] = f32_to_bf16(p[0]) | (f32_to_bf16(p[1]) << 16);
  }
  return __builtin_bit_cast(v16bf, au);
}

// Prepacked B fragments: flat index ((kt*8+nt)*32+lane)*16+e  (bf16 elems).
// Works for global (global_load_b128) or LDS-resident (ds_load_b128) source.
__device__ __forceinline__ v16bf load_bfrag(const unsigned short* wp,
                                            int kt, int nt, int lane) {
  const v8u* p = (const v8u*)(wp + ((((kt << 3) + nt) * 32 + lane) << 4));
  return __builtin_bit_cast(v16bf, *p);
}

// ---------------------------------------------------------------- kernels

// Repack W[k][n] (f32 row-major 128x128) -> bf16 WMMA B-fragment order.
// B layout: lane l: n = nt*16 + (l&15); K = kt*32 + (e<8 ? (l>>4)*8 + e : 16 + (l>>4)*8 + e-8)
__global__ void __launch_bounds__(256)
prepack_weights(const float* __restrict__ WA, const float* __restrict__ WB,
                const float* __restrict__ WC, const float* __restrict__ WD,
                const float* __restrict__ WE, unsigned short* __restrict__ wpk) {
  int idx = blockIdx.x * blockDim.x + threadIdx.x;
  if (idx >= 5 * 16384) return;
  int w    = idx >> 14;
  int rest = idx & 16383;
  int e    = rest & 15;
  int lane = (rest >> 4) & 31;
  int nt   = (rest >> 9) & 7;
  int kt   = rest >> 12;
  int n = nt * 16 + (lane & 15);
  int k = kt * 32 + ((e < 8) ? 0 : 8) + (lane >> 4) * 8 + e;
  const float* W = (w == 0) ? WA : (w == 1) ? WB : (w == 2) ? WC : (w == 3) ? WD : WE;
  wpk[idx] = (unsigned short)f32_to_bf16(W[k * Dk + n]);
}

// Ah,Bh,Dh,Eh = h @ W{A,B,D,E} + b  -- one 16x128 strip per wave.
// Accumulators start at inline 0 (ISA: SRC2 may be inline 0); bias added at store.
__global__ void __launch_bounds__(256)
node_gemm(const float* __restrict__ h, const unsigned short* __restrict__ wpk,
          const float* __restrict__ bA, const float* __restrict__ bB,
          const float* __restrict__ bD, const float* __restrict__ bE,
          float* __restrict__ Ah, float* __restrict__ Bh,
          float* __restrict__ Dh, float* __restrict__ Eh) {
  const int tid  = threadIdx.x;
  const int lane = tid & 31, l15 = lane & 15, hi = lane >> 4;
  const int strip = blockIdx.x * 8 + (tid >> 5);
  if (strip * 16 >= NN) return;                 // wave-uniform
  const int mbase = strip * 16;
  const int m = mbase + l15;

  v16bf a0 = load_afrag(h, m, 0, hi);
  v16bf a1 = load_afrag(h, m, 1, hi);
  v16bf a2 = load_afrag(h, m, 2, hi);
  v16bf a3 = load_afrag(h, m, 3, hi);

  const int    wslot[4]  = {0, 1, 3, 4};        // WA, WB, WD, WE slots in wpk
  const float* biases[4] = {bA, bB, bD, bE};
  float*       outs[4]   = {Ah, Bh, Dh, Eh};

#pragma unroll
  for (int wi = 0; wi < 4; ++wi) {
    const unsigned short* wp = wpk + wslot[wi] * 16384;
    const float* bias = biases[wi];
    float* out = outs[wi];
#pragma unroll
    for (int nt = 0; nt < 8; ++nt) {
      const int n = nt * 16 + l15;
      v8f acc = {0.f, 0.f, 0.f, 0.f, 0.f, 0.f, 0.f, 0.f};
      acc = __builtin_amdgcn_wmma_f32_16x16x32_bf16(false, a0, false, load_bfrag(wp, 0, nt, lane), (short)0, acc, false, false);
      acc = __builtin_amdgcn_wmma_f32_16x16x32_bf16(false, a1, false, load_bfrag(wp, 1, nt, lane), (short)0, acc, false, false);
      acc = __builtin_amdgcn_wmma_f32_16x16x32_bf16(false, a2, false, load_bfrag(wp, 2, nt, lane), (short)0, acc, false, false);
      acc = __builtin_amdgcn_wmma_f32_16x16x32_bf16(false, a3, false, load_bfrag(wp, 3, nt, lane), (short)0, acc, false, false);
      const float bv = bias[n];
#pragma unroll
      for (int r = 0; r < 8; ++r)
        out[(size_t)(mbase + hi * 8 + r) * Dk + n] = acc[r] + bv;
    }
  }
}

// Fused: Ce = e@WC + bC; e_ij = Ce + Dh[src] + Eh[dst]; sigmoid; scatter-add
// num/den/deg; store e_ij; accumulate per-column BN stats in registers.
// WC fragments staged once per block into 32KB LDS (amortized over ~100 strips).
__global__ void __launch_bounds__(256)
edge_gemm_epilogue(const float* __restrict__ e, const int* __restrict__ src,
                   const int* __restrict__ dst, const unsigned short* __restrict__ wpk,
                   const float* __restrict__ bC,
                   const float* __restrict__ Bh, const float* __restrict__ Dh,
                   const float* __restrict__ Eh,
                   float* __restrict__ num, float* __restrict__ den,
                   float* __restrict__ deg, float* __restrict__ eij,
                   float* __restrict__ stats) {
  __shared__ unsigned short ldsWC[16384];       // 32 KB, fragment-ordered WC

  const int tid   = threadIdx.x;
  const int lane  = tid & 31, l15 = lane & 15, hi = lane >> 4;
  const int waveId = (blockIdx.x * blockDim.x + tid) >> 5;
  const int nWaves = (gridDim.x * blockDim.x) >> 5;

  {  // cooperative stage: 2048 x uint4 (16B) = 32KB
    const uint4* g = (const uint4*)(wpk + 2 * 16384);  // WC slot
    uint4* l = (uint4*)ldsWC;
#pragma unroll
    for (int i = 0; i < 8; ++i)
      l[tid + i * 256] = g[tid + i * 256];
  }
  __syncthreads();

  float es[8], ess[8];
#pragma unroll
  for (int i = 0; i < 8; ++i) { es[i] = 0.f; ess[i] = 0.f; }

  const int nStrips = EE / 16;                  // 50000, exact
  for (int strip = waveId; strip < nStrips; strip += nWaves) {
    // prefetch next strip's activation rows (4 x 128B lines per 512B row)
    const int nextStrip = strip + nWaves;
    if (nextStrip < nStrips) {
      const float* pf = e + (size_t)(nextStrip * 16 + l15) * Dk + hi * 64;
      __builtin_prefetch(pf, 0, 0);
      __builtin_prefetch(pf + 32, 0, 0);
    }

    const int mbase = strip * 16;
    const int m = mbase + l15;
    v16bf a0 = load_afrag(e, m, 0, hi);
    v16bf a1 = load_afrag(e, m, 1, hi);
    v16bf a2 = load_afrag(e, m, 2, hi);
    v16bf a3 = load_afrag(e, m, 3, hi);

    int s[8], d[8];
#pragma unroll
    for (int r = 0; r < 8; ++r) {
      s[r] = src[mbase + hi * 8 + r];
      d[r] = dst[mbase + hi * 8 + r];
    }

#pragma unroll
    for (int nt = 0; nt < 8; ++nt) {
      const int n = nt * 16 + l15;
      v8f acc = {0.f, 0.f, 0.f, 0.f, 0.f, 0.f, 0.f, 0.f};
      acc = __builtin_amdgcn_wmma_f32_16x16x32_bf16(false, a0, false, load_bfrag(ldsWC, 0, nt, lane), (short)0, acc, false, false);
      acc = __builtin_amdgcn_wmma_f32_16x16x32_bf16(false, a1, false, load_bfrag(ldsWC, 1, nt, lane), (short)0, acc, false, false);
      acc = __builtin_amdgcn_wmma_f32_16x16x32_bf16(false, a2, false, load_bfrag(ldsWC, 2, nt, lane), (short)0, acc, false, false);
      acc = __builtin_amdgcn_wmma_f32_16x16x32_bf16(false, a3, false, load_bfrag(ldsWC, 3, nt, lane), (short)0, acc, false, false);
      const float bv = bC[n];

#pragma unroll
      for (int r = 0; r < 8; ++r) {
        const size_t so  = (size_t)s[r] * Dk + n;
        const size_t dof = (size_t)d[r] * Dk + n;
        float cij = acc[r] + bv + Dh[so] + Eh[dof];     // L2-resident gathers
        float sig = 1.0f / (1.0f + __expf(-cij));
        atomicAdd(&num[dof], sig * Bh[so]);             // L2 atomics
        atomicAdd(&den[dof], sig);
        eij[(size_t)(mbase + hi * 8 + r) * Dk + n] = cij;
        es[nt]  += cij;
        ess[nt] += cij * cij;
      }
      if (nt == 0 && l15 == 0) {                        // lanes 0 & 16 own 8 rows each
#pragma unroll
        for (int r = 0; r < 8; ++r) atomicAdd(&deg[d[r]], 1.0f);
      }
    }
  }

  float* esum   = stats + 256;
  float* esumsq = stats + 384;
#pragma unroll
  for (int nt = 0; nt < 8; ++nt) {
    int n = nt * 16 + l15;
    atomicAdd(&esum[n], es[nt]);
    atomicAdd(&esumsq[n], ess[nt]);
  }
}

// h_agg = where(deg>0, Ah + num/den', h); column BN stats.
__global__ void __launch_bounds__(256)
hagg_stats(const float* __restrict__ h, const float* __restrict__ Ah,
           const float* __restrict__ num, const float* __restrict__ den,
           const float* __restrict__ deg, float* __restrict__ hagg,
           float* __restrict__ stats) {
  const int tid = threadIdx.x;
  const int col = tid & 127;
  const int rg  = tid >> 7;
  const int mbase = blockIdx.x * 64;
  float s = 0.f, ss = 0.f;
  for (int i = 0; i < 32; ++i) {
    int m = mbase + i * 2 + rg;
    if (m >= NN) break;
    size_t off = (size_t)m * Dk + col;
    float dn = den[off];
    float v = Ah[off] + num[off] / (dn == 0.f ? 1.f : dn);
    if (deg[m] <= 0.f) v = h[off];
    hagg[off] = v;
    s += v; ss += v * v;
  }
  atomicAdd(&stats[col], s);
  atomicAdd(&stats[128 + col], ss);
}

// Turn (sum, sumsq) into per-column (scale, shift) for both BNs.
__global__ void finalize_stats(const float* __restrict__ gh, const float* __restrict__ bh,
                               const float* __restrict__ ge, const float* __restrict__ be,
                               float* __restrict__ stats) {
  int c = threadIdx.x;
  if (c >= 128) return;
  float mh = stats[c] / (float)NN;
  float vh = stats[128 + c] / (float)NN - mh * mh;
  float sch = gh[c] * rsqrtf(vh + BN_EPS);
  stats[512 + c] = sch;
  stats[640 + c] = bh[c] - mh * sch;
  float me = stats[256 + c] / (float)EE;
  float ve = stats[384 + c] / (float)EE - me * me;
  float sce = ge[c] * rsqrtf(ve + BN_EPS);
  stats[768 + c] = sce;
  stats[896 + c] = be[c] - me * sce;
}

// out = x + relu(t*scale + shift), float4-vectorized (128 cols = 32 float4).
__global__ void __launch_bounds__(256)
write_out(const float* __restrict__ x, const float* __restrict__ t,
          const float* __restrict__ stats, int scaleOff,
          float* __restrict__ out, long long n4) {
  long long i = (long long)blockIdx.x * blockDim.x + threadIdx.x;
  if (i >= n4) return;
  int c4 = (int)(i & 31);
  float4 sc = ((const float4*)(stats + scaleOff))[c4];
  float4 sh = ((const float4*)(stats + scaleOff + 128))[c4];
  float4 xv = ((const float4*)x)[i];
  float4 tv = ((const float4*)t)[i];
  float4 o;
  o.x = xv.x + fmaxf(tv.x * sc.x + sh.x, 0.f);
  o.y = xv.y + fmaxf(tv.y * sc.y + sh.y, 0.f);
  o.z = xv.z + fmaxf(tv.z * sc.z + sh.z, 0.f);
  o.w = xv.w + fmaxf(tv.w * sc.w + sh.w, 0.f);
  ((float4*)out)[i] = o;
}

// ---------------------------------------------------------------- launch

extern "C" void kernel_launch(void* const* d_in, const int* in_sizes, int n_in,
                              void* d_out, int out_size, void* d_ws, size_t ws_size,
                              hipStream_t stream) {
  (void)in_sizes; (void)n_in; (void)out_size; (void)ws_size;
  const float* h  = (const float*)d_in[0];
  const float* e  = (const float*)d_in[1];
  const int*   src = (const int*)d_in[2];
  const int*   dst = (const int*)d_in[3];
  const float* WA = (const float*)d_in[4];
  const float* bA = (const float*)d_in[5];
  const float* WB = (const float*)d_in[6];
  const float* bB = (const float*)d_in[7];
  const float* WC = (const float*)d_in[8];
  const float* bC = (const float*)d_in[9];
  const float* WD = (const float*)d_in[10];
  const float* bD = (const float*)d_in[11];
  const float* WE = (const float*)d_in[12];
  const float* bE = (const float*)d_in[13];
  const float* gamma_h = (const float*)d_in[14];
  const float* beta_h  = (const float*)d_in[15];
  const float* gamma_e = (const float*)d_in[16];
  const float* beta_e  = (const float*)d_in[17];

  char* ws = (char*)d_ws;
  size_t off = 0;
  unsigned short* wpk = (unsigned short*)(ws + off); off += (size_t)5 * 16384 * 2;
  off = (off + 255) & ~(size_t)255;
  const size_t nd = (size_t)NN * Dk * sizeof(float);
  float* Ah  = (float*)(ws + off); off += nd;
  float* Bh  = (float*)(ws + off); off += nd;
  float* Dh  = (float*)(ws + off); off += nd;
  float* Eh  = (float*)(ws + off); off += nd;
  float* num = (float*)(ws + off); off += nd;
  float* den = (float*)(ws + off); off += nd;
  float* hagg = (float*)(ws + off); off += nd;
  float* deg = (float*)(ws + off); off += (size_t)NN * sizeof(float);
  off = (off + 255) & ~(size_t)255;
  float* stats = (float*)(ws + off); off += 1024 * sizeof(float);
  off = (off + 255) & ~(size_t)255;
  float* eij = (float*)(ws + off); off += (size_t)EE * Dk * sizeof(float);

  hipMemsetAsync(num, 0, nd, stream);
  hipMemsetAsync(den, 0, nd, stream);
  hipMemsetAsync(deg, 0, (size_t)NN * sizeof(float), stream);
  hipMemsetAsync(stats, 0, 512 * sizeof(float), stream);

  prepack_weights<<<(5 * 16384 + 255) / 256, 256, 0, stream>>>(WA, WB, WC, WD, WE, wpk);

  node_gemm<<<(NN / 16 + 7) / 8, 256, 0, stream>>>(h, wpk, bA, bB, bD, bE, Ah, Bh, Dh, Eh);

  edge_gemm_epilogue<<<512, 256, 0, stream>>>(e, src, dst, wpk, bC, Bh, Dh, Eh,
                                              num, den, deg, eij, stats);

  hagg_stats<<<(NN + 63) / 64, 256, 0, stream>>>(h, Ah, num, den, deg, hagg, stats);

  finalize_stats<<<1, 128, 0, stream>>>(gamma_h, beta_h, gamma_e, beta_e, stats);

  float* out = (float*)d_out;
  long long hn4 = (long long)NN * 32;
  long long en4 = (long long)EE * 32;
  write_out<<<(int)((hn4 + 255) / 256), 256, 0, stream>>>(h, hagg, stats, 512, out, hn4);
  write_out<<<(int)((en4 + 255) / 256), 256, 0, stream>>>(e, eij, stats, 768,
                                                          out + (size_t)NN * Dk, en4);
}
